// CLARM_56693568307877
// MI455X (gfx1250) — matrix operand order repaired and checked by reading, not compiled, not run on hardware
//
#include <hip/hip_runtime.h>

#define HIDDEN 640
#define NPOS   196      // 14*14 spatial positions
#define NPAD   224      // padded to 7*32 for K-steps of rec matmul
#define DINNER 80
#define DPAD   96       // padded to 3*32 for K-steps of score matmul
#define NB_A   64
#define NB_B   128
#define NTILES 13       // ceil(196/16) row tiles
#define PITCH  228      // LDS row pitch in floats (224 + 4 pad)
#define CHUNK_NT 4      // row tiles per LDS chunk (64 rows, ~58KB LDS)
#define XPITCH (HIDDEN + 8)   // bf16 X-slab row pitch (648 elems = 1296B, 16B aligned)
#define WROWS  256      // W rows padded 240 -> 256 (16 i-tiles, uniform wave work)

typedef __bf16 bf16x16 __attribute__((ext_vector_type(16)));
typedef __bf16 bf16x8  __attribute__((ext_vector_type(8)));
typedef float  f32x8   __attribute__((ext_vector_type(8)));
typedef unsigned int u32x4 __attribute__((ext_vector_type(4)));
typedef int    i32x4  __attribute__((ext_vector_type(4)));
typedef int    i32x8  __attribute__((ext_vector_type(8)));

__device__ __forceinline__ bf16x16 join16(bf16x8 lo, bf16x8 hi) {
  union { bf16x16 v; bf16x8 h[2]; } u;
  u.h[0] = lo; u.h[1] = hi;
  return u.v;
}

// ---------------------------------------------------------------------------
// Kernel 0: zero workspace region (padding columns/rows must be 0 for WMMA)
// ---------------------------------------------------------------------------
__global__ void CLARM_zero(uint32_t* __restrict__ p, size_t nwords) {
  size_t i = (size_t)blockIdx.x * blockDim.x + threadIdx.x;
  size_t stride = (size_t)gridDim.x * blockDim.x;
  for (; i < nwords; i += stride) p[i] = 0u;
}

// ---------------------------------------------------------------------------
// Kernel 1: convert W_qkv [240][640] f32 -> bf16 (rows 240..255 stay zero)
// ---------------------------------------------------------------------------
__global__ void CLARM_convw(const float* __restrict__ W, __bf16* __restrict__ Wb, int n) {
  int i = blockIdx.x * blockDim.x + threadIdx.x;
  if (i < n) Wb[i] = (__bf16)W[i];
}

// ---------------------------------------------------------------------------
// Kernel 2: QKV projection.
// feat layout [item][640][196] (f32).  out[n][i] = sum_h x[n][h] * W[i][h]
// Block = (n-tile, item). TDM DMAs the strided [640h x 16n] f32 tile into
// LDS (one descriptor, TENSORcnt-tracked), threads convert once into a
// fragment-major bf16 slab, then 20 K-steps of WMMA; every wave handles
// exactly 2 i-tiles (W padded to 256 rows) -> no exec-mask games, WMMA
// always runs with full EXEC as the ISA requires. Tile 15 is never stored.
// Stores q (pre-scaled by 1/sqrt(80)), k, v as [n:224][d:96] bf16 and
// vT as [d:96][n:224] bf16 (so rec matmul B-fragments are contiguous).
// ---------------------------------------------------------------------------
__global__ void CLARM_qkv(const float* __restrict__ feat, const __bf16* __restrict__ Wb,
                          __bf16* __restrict__ q, __bf16* __restrict__ k,
                          __bf16* __restrict__ v, __bf16* __restrict__ vT,
                          float qscale)
{
  const int item = blockIdx.y;
  const int n0   = blockIdx.x * 16;
  const int tid  = threadIdx.x;
  const int wave = tid >> 5;
  const int lane = tid & 31;
  const int half = lane >> 4;
  const int lr   = lane & 15;

  __shared__ float  Xf[HIDDEN * 16];        // [h][16n] f32 tile from TDM (40KB)
  __shared__ __bf16 Xs[16][XPITCH];         // [n][h] bf16, fragment-major (20.7KB)

  const float* fbase = feat + (size_t)item * HIDDEN * NPOS + n0;

  // ---- TDM: DMA the 2D tile (tile_dim0=16 cols, tile_dim1=640 rows,
  //      row stride 196 f32) from global into LDS; wave 0 issues & waits ----
  if (wave == 0) {
    uint64_t ga  = (uint64_t)(uintptr_t)fbase;
    uint32_t lds = (uint32_t)(uintptr_t)&Xf[0];   // LDS offset = addr[31:0]
    int dim0 = NPOS - n0;                          // valid cols; OOB cols -> 0
    if (dim0 < 1) dim0 = 1;

    u32x4 g0;
    g0[0] = 1u;                                          // count=1, no gather
    g0[1] = lds;                                         // lds_addr
    g0[2] = (uint32_t)ga;                                // global_addr[31:0]
    g0[3] = (uint32_t)((ga >> 32) & 0x1FFFFFFu)          // global_addr[56:32]
            | (2u << 30);                                // type=2 (image)
    i32x8 g1;
    g1[0] = (int)(2u << 16);                             // data_size=4B
    g1[1] = (int)((uint32_t)dim0 << 16);                 // tensor_dim0[15:0]
    g1[2] = (int)(((uint32_t)dim0 >> 16)                 // tensor_dim0[31:16]
            | ((uint32_t)HIDDEN << 16));                 // tensor_dim1[15:0]
    g1[3] = (int)(((uint32_t)HIDDEN >> 16)               // tensor_dim1[31:16]
            | (16u << 16));                              // tile_dim0=16
    g1[4] = (int)(HIDDEN);                               // tile_dim1=640, tile_dim2=0
    g1[5] = (int)(NPOS);                                 // tensor_dim0_stride lo32
    g1[6] = 0;                                           // stride0 hi / stride1 lo
    g1[7] = 0;
    i32x4 g2 = {}; i32x4 g3 = {};
#if __clang_major__ >= 23
    i32x8 g1b = {};
    __builtin_amdgcn_tensor_load_to_lds(g0, g1, g2, g3, g1b, 0);
#else
    __builtin_amdgcn_tensor_load_to_lds(g0, g1, g2, g3, 0);
#endif
    __builtin_amdgcn_s_wait_tensorcnt(0);
  }
  __syncthreads();

  // ---- Convert to fragment-major bf16 slab (guard n>=196 -> 0) ----
  for (int e = tid; e < HIDDEN * 16; e += 256) {
    int hh = e >> 4, nn = e & 15;
    float val = (n0 + nn < NPOS) ? Xf[e] : 0.0f;
    Xs[nn][hh] = (__bf16)val;
  }
  __syncthreads();

  // ---- K-loop: 20 steps of 32h, accumulators resident, uniform EXEC ----
  f32x8 acc0 = {}; f32x8 acc1 = {};
  const int it0 = wave;       // i-tile #1 (0..7)
  const int it1 = wave + 8;   // i-tile #2 (8..15; tile 15 computes zeros)

  for (int h0 = 0; h0 < HIDDEN; h0 += 32) {
    // A-fragment: two contiguous b128 LDS reads per lane (ISA 16x32 layout)
    const __bf16* xr = &Xs[lr][h0 + half * 8];
    bf16x16 a = join16(*(const bf16x8*)xr, *(const bf16x8*)(xr + 16));
    {
      const __bf16* wb = Wb + (size_t)(it0 * 16 + lr) * HIDDEN + h0 + half * 8;
      bf16x16 b = join16(*(const bf16x8*)wb, *(const bf16x8*)(wb + 16));
      acc0 = __builtin_amdgcn_wmma_f32_16x16x32_bf16(false, a, false, b,
                                                     (short)0, acc0, false, false);
    }
    {
      const __bf16* wb = Wb + (size_t)(it1 * 16 + lr) * HIDDEN + h0 + half * 8;
      bf16x16 b = join16(*(const bf16x8*)wb, *(const bf16x8*)(wb + 16));
      acc1 = __builtin_amdgcn_wmma_f32_16x16x32_bf16(false, a, false, b,
                                                     (short)0, acc1, false, false);
    }
  }

  // ---- Write out tiles. i-tile<5 -> q (scaled), <10 -> k, <15 -> v (+vT) ----
  const size_t base = (size_t)item * ((size_t)NPAD * DPAD);
#pragma unroll 1
  for (int which = 0; which < 2; ++which) {
    int it = which ? it1 : it0;
    if (it >= 15) break;                     // padding tile: computed, not stored
    f32x8 acc = which ? acc1 : acc0;
    __bf16* arr = (it < 5) ? q : (it < 10 ? k : v);
    float s = (it < 5) ? qscale : 1.0f;
    int dcol = (it % 5) * 16 + lr;
#pragma unroll
    for (int j = 0; j < 8; ++j) {
      int n = n0 + j + half * 8;              // D layout: row = j + 8*(lane>=16)
      float val = acc[j] * s;
      arr[base + (size_t)n * DPAD + dcol] = (__bf16)val;
      if (it >= 10) vT[base + (size_t)dcol * NPAD + n] = (__bf16)val;
    }
  }
}

// ---------------------------------------------------------------------------
// Kernel 3: fused attention + similarity. One block per (b, a) pair.
// Chunked over 64 query rows so LDS stays ~58KB (multiple blocks per WGP).
// ---------------------------------------------------------------------------
__global__ void CLARM_attn(const __bf16* __restrict__ qB, const __bf16* __restrict__ kA,
                           const __bf16* __restrict__ vAT, const __bf16* __restrict__ vB,
                           float* __restrict__ out)
{
  __shared__ float S[CHUNK_NT * 16 * PITCH];   // score/prob chunk
  __shared__ float red[8];

  const int a = blockIdx.x;      // support item  0..63
  const int b = blockIdx.y;      // query item    0..127
  const int tid  = threadIdx.x;
  const int wave = tid >> 5;
  const int lane = tid & 31;
  const int half = lane >> 4;
  const int lr   = lane & 15;

  const size_t per = (size_t)NPAD * DPAD;
  const __bf16* q  = qB  + (size_t)b * per;   // [224][96], pre-scaled
  const __bf16* k  = kA  + (size_t)a * per;   // [224][96]
  const __bf16* vT = vAT + (size_t)a * per;   // [96][224]
  const __bf16* vb = vB  + (size_t)b * per;   // [224][96]

  float local = 0.0f;

  for (int c0 = 0; c0 < NTILES; c0 += CHUNK_NT) {
    const int ntc = (NTILES - c0 < CHUNK_NT) ? (NTILES - c0) : CHUNK_NT;

    // ---- Phase 1: score tiles (q_b . k_a^T), f32 into LDS ----
    for (int t = wave; t < ntc * NTILES; t += 8) {
      int ntl = t / NTILES, mt = t % NTILES;
      int n0g = (c0 + ntl) * 16;
      int n0l = ntl * 16;
      int m0  = mt * 16;
      if (mt + 1 < NTILES)
        __builtin_prefetch(k + (size_t)((mt + 1) * 16 + lr) * DPAD, 0, 0);
      f32x8 acc = {};
#pragma unroll
      for (int ks = 0; ks < 3; ++ks) {
        const __bf16* pa = q + (size_t)(n0g + lr) * DPAD + ks * 32 + half * 8;
        bf16x16 afr = join16(*(const bf16x8*)pa, *(const bf16x8*)(pa + 16));
        const __bf16* pb = k + (size_t)(m0 + lr) * DPAD + ks * 32 + half * 8;
        bf16x16 bfr = join16(*(const bf16x8*)pb, *(const bf16x8*)(pb + 16));
        acc = __builtin_amdgcn_wmma_f32_16x16x32_bf16(false, afr, false, bfr,
                                                      (short)0, acc, false, false);
      }
#pragma unroll
      for (int j = 0; j < 8; ++j)
        S[(n0l + j + half * 8) * PITCH + m0 + lr] = acc[j];
    }
    __syncthreads();

    // ---- Phase 2: masked row softmax, bf16 probs written in place ----
    for (int rl = wave; rl < ntc * 16; rl += 8) {
      int r = c0 * 16 + rl;                 // global query row
      float* row = S + rl * PITCH;
      __bf16* prow = (__bf16*)row;          // bf16 row occupies first half
      if (r < NPOS) {
        float vals[7];
        float mx = -3.4e38f;
#pragma unroll
        for (int c = 0; c < 7; ++c) {
          int m = lane + c * 32;
          float x = (m < NPOS) ? row[m] : -3.4e38f;
          vals[c] = x;
          mx = fmaxf(mx, x);
        }
#pragma unroll
        for (int o = 16; o > 0; o >>= 1) mx = fmaxf(mx, __shfl_xor(mx, o, 32));
        float sum = 0.0f;
#pragma unroll
        for (int c = 0; c < 7; ++c) {
          int m = lane + c * 32;
          float e = (m < NPOS) ? __expf(vals[c] - mx) : 0.0f;
          vals[c] = e;
          sum += e;
        }
#pragma unroll
        for (int o = 16; o > 0; o >>= 1) sum += __shfl_xor(sum, o, 32);
        float inv = 1.0f / sum;
#pragma unroll
        for (int c = 0; c < 7; ++c) {
          int m = lane + c * 32;
          prow[m] = (__bf16)(vals[c] * inv);   // m in [196,224) gets 0
        }
      } else {
        for (int m = lane; m < NPAD; m += 32) prow[m] = (__bf16)0.0f;
      }
    }
    __syncthreads();

    // ---- Phase 3: rec = probs @ v_a, accumulate (v_b - rec)^2 ----
    for (int t = wave; t < ntc * 5; t += 8) {
      int ntl = t / 5, dt = t % 5;
      int n0l = ntl * 16;
      int d0  = dt * 16;
      f32x8 acc = {};
#pragma unroll
      for (int ks = 0; ks < 7; ++ks) {
        const __bf16* pa = (const __bf16*)(S + (size_t)(n0l + lr) * PITCH)
                           + ks * 32 + half * 8;
        bf16x16 afr = join16(*(const bf16x8*)pa, *(const bf16x8*)(pa + 16));
        const __bf16* pb = vT + (size_t)(d0 + lr) * NPAD + ks * 32 + half * 8;
        bf16x16 bfr = join16(*(const bf16x8*)pb, *(const bf16x8*)(pb + 16));
        acc = __builtin_amdgcn_wmma_f32_16x16x32_bf16(false, afr, false, bfr,
                                                      (short)0, acc, false, false);
      }
#pragma unroll
      for (int j = 0; j < 8; ++j) {
        int n = (c0 + ntl) * 16 + j + half * 8;
        int d = d0 + lr;                    // always < 80 (dt < 5)
        if (n < NPOS) {
          float diff = (float)vb[(size_t)n * DPAD + d] - acc[j];
          local += diff * diff;
        }
      }
    }
    __syncthreads();
  }

  // ---- Block reduction, write -sum ----
#pragma unroll
  for (int o = 16; o > 0; o >>= 1) local += __shfl_xor(local, o, 32);
  if (lane == 0) red[wave] = local;
  __syncthreads();
  if (tid == 0) {
    float tot = 0.0f;
#pragma unroll
    for (int w = 0; w < 8; ++w) tot += red[w];
    out[(size_t)b * NB_A + a] = -tot;
  }
}

// ---------------------------------------------------------------------------
extern "C" void kernel_launch(void* const* d_in, const int* in_sizes, int n_in,
                              void* d_out, int out_size, void* d_ws, size_t ws_size,
                              hipStream_t stream) {
  (void)in_sizes; (void)n_in; (void)out_size; (void)ws_size;

  const float* feats_a = (const float*)d_in[0];   // [64, 640, 14, 14]
  const float* feats_b = (const float*)d_in[1];   // [128, 640, 14, 14]
  const float* Wqkv    = (const float*)d_in[2];   // [240, 640]
  float* out = (float*)d_out;                     // [128, 64]

  const size_t per = (size_t)NPAD * DPAD;                  // 21504 elems/item
  const size_t arr = per * (size_t)(NB_A + NB_B);          // elems per array
  char* ws = (char*)d_ws;
  __bf16* Wb = (__bf16*)ws;                                // [256][640] bf16
  __bf16* qq = (__bf16*)(ws + (size_t)WROWS * HIDDEN * 2); // 327680 B
  __bf16* kk = qq + arr;
  __bf16* vv = kk + arr;
  __bf16* vT = vv + arr;

  const float qscale = 0.11180339887498949f;               // 1/sqrt(80)

  // Zero Wb (incl. pad rows 240..255) and q/k/v/vT padding in one pass
  size_t nwords = ((size_t)WROWS * HIDDEN * 2 + 4 * arr * 2) / 4;
  CLARM_zero<<<2048, 256, 0, stream>>>((uint32_t*)ws, nwords);

  // Convert W to bf16 (rows 0..239)
  CLARM_convw<<<(240 * HIDDEN + 255) / 256, 256, 0, stream>>>(Wqkv, Wb, 240 * HIDDEN);

  // QKV projection: support set (items 0..63), then query set (items 64..191)
  CLARM_qkv<<<dim3(NPAD / 16, NB_A), 256, 0, stream>>>(
      feats_a, Wb, qq, kk, vv, vT, qscale);
  CLARM_qkv<<<dim3(NPAD / 16, NB_B), 256, 0, stream>>>(
      feats_b, Wb, qq + (size_t)NB_A * per, kk + (size_t)NB_A * per,
      vv + (size_t)NB_A * per, vT + (size_t)NB_A * per, qscale);

  // Fused attention + similarity: one block per (b, a) pair
  CLARM_attn<<<dim3(NB_A, NB_B), 256, 0, stream>>>(
      qq + (size_t)NB_A * per,   // q of query items
      kk,                        // k of support items
      vT,                        // v^T of support items
      vv + (size_t)NB_A * per,   // v of query items
      out);
}
